// MultiHeadAttention_23089744183625
// MI455X (gfx1250) — matrix operand order repaired and checked by reading, not compiled, not run on hardware
//
#include <hip/hip_runtime.h>
#include <hip/hip_bf16.h>

// MHA forward for B=4, S=2048, H=1024, NH=16, DK=64 on gfx1250 (MI455X).
// bf16 WMMA for all four matmul stages, flash-attention inner loop, f32 accum.
// GEMMs stage weight tiles in LDS via GLOBAL_LOAD_ASYNC_TO_LDS (double-
// buffered, s_wait_asynccnt) when the toolchain exposes the builtins.

#define B_  4
#define S_  2048
#define H_  1024
#define NH_ 16
#define DK_ 64
#define NEG_BIG (-1.0e30f)

#ifndef __has_builtin
#define __has_builtin(x) 0
#endif
#if __has_builtin(__builtin_amdgcn_global_load_async_to_lds_b128) && \
    __has_builtin(__builtin_amdgcn_s_wait_asynccnt)
#define USE_ASYNC_LDS 1
#else
#define USE_ASYNC_LDS 0
#endif

#define AS1 __attribute__((address_space(1)))
#define AS3 __attribute__((address_space(3)))

typedef __bf16 bf16_t;
typedef __attribute__((ext_vector_type(16))) __bf16        v16bf;
typedef __attribute__((ext_vector_type(8)))  float         v8f;
typedef __attribute__((ext_vector_type(4))) unsigned int   u32x4;
typedef __attribute__((ext_vector_type(4)))  float         f32x4;
typedef int vi4 __attribute__((vector_size(16)));  // matches builtin's V4i

union FragBF { v16bf v; u32x4 u[2]; };

// ---- fragment loaders (layouts per CDNA5 ISA 7.12.2) --------------------

// A-matrix 16x32 bf16: lane = row (lane&15); lane half selects K offset 0/8;
// elements 0..7 -> k = klo+j, elements 8..15 -> k = 16+klo+j.
__device__ __forceinline__ v16bf load_matA_bf16(const bf16_t* base, int ld,
                                                int kbase, int lane) {
  int r   = lane & 15;
  int klo = (lane >> 4) << 3;
  const bf16_t* p = base + (size_t)r * ld + kbase + klo;
  FragBF f;
  f.u[0] = *(const u32x4*)(p);
  f.u[1] = *(const u32x4*)(p + 16);
  return f.v;
}

// Same A layout but source is f32; convert to bf16 on the fly.
__device__ __forceinline__ v16bf load_matA_f32(const float* base, int ld,
                                               int kbase, int lane) {
  int r   = lane & 15;
  int klo = (lane >> 4) << 3;
  const float* p = base + (size_t)r * ld + kbase + klo;
  f32x4 x0 = *(const f32x4*)(p);
  f32x4 x1 = *(const f32x4*)(p + 4);
  f32x4 x2 = *(const f32x4*)(p + 16);
  f32x4 x3 = *(const f32x4*)(p + 20);
  v16bf v;
  v[0]=(bf16_t)x0.x; v[1]=(bf16_t)x0.y; v[2]=(bf16_t)x0.z; v[3]=(bf16_t)x0.w;
  v[4]=(bf16_t)x1.x; v[5]=(bf16_t)x1.y; v[6]=(bf16_t)x1.z; v[7]=(bf16_t)x1.w;
  v[8]=(bf16_t)x2.x; v[9]=(bf16_t)x2.y; v[10]=(bf16_t)x2.z; v[11]=(bf16_t)x2.w;
  v[12]=(bf16_t)x3.x; v[13]=(bf16_t)x3.y; v[14]=(bf16_t)x3.z; v[15]=(bf16_t)x3.w;
  return v;
}

// B-matrix 32x16 bf16 taken as rows of a "transposed" row-major operand:
// lane = output column n (= row n of the stored matrix); lane half selects
// K offset 0/8; elements 0..7 -> k = klo+j, elements 8..15 -> k = 16+klo+j.
__device__ __forceinline__ v16bf load_matB_bf16(const bf16_t* base, int ld,
                                                int kbase, int lane) {
  int r  = lane & 15;
  int kh = (lane >> 4) << 3;
  const bf16_t* p = base + (size_t)r * ld + kbase + kh;
  FragBF f;
  f.u[0] = *(const u32x4*)(p);
  f.u[1] = *(const u32x4*)(p + 16);
  return f.v;
}

// B fragment from an LDS-staged 64x32 bf16 tile (row stride 32 elements).
__device__ __forceinline__ v16bf load_matB_lds(const bf16_t* tile, int lane) {
  int r  = lane & 15;
  int kh = (lane >> 4) << 3;
  const bf16_t* p = tile + r * 32 + kh;
  FragBF f;
  f.u[0] = *(const u32x4*)(p);
  f.u[1] = *(const u32x4*)(p + 16);
  return f.v;
}

__device__ __forceinline__ v8f wmma_bf16(v16bf a, v16bf b, v8f c) {
  return __builtin_amdgcn_wmma_f32_16x16x32_bf16(false, a, false, b,
                                                 (short)0, c, false, false);
}

__device__ __forceinline__ float redmax16(float v) {
  v = fmaxf(v, __shfl_xor(v, 1, 16));
  v = fmaxf(v, __shfl_xor(v, 2, 16));
  v = fmaxf(v, __shfl_xor(v, 4, 16));
  v = fmaxf(v, __shfl_xor(v, 8, 16));
  return v;
}
__device__ __forceinline__ float redsum16(float v) {
  v += __shfl_xor(v, 1, 16);
  v += __shfl_xor(v, 2, 16);
  v += __shfl_xor(v, 4, 16);
  v += __shfl_xor(v, 8, 16);
  return v;
}

// Stage a 64(N) x 32(K) bf16 weight tile into LDS. W is pre-offset to the
// tile's first row; 256 16B chunks, 128 threads x 2 chunks.
__device__ __forceinline__ void stage_w_tile(const bf16_t* __restrict__ W,
                                             int kbase, bf16_t* tile, int tid) {
#pragma unroll
  for (int c = tid; c < 256; c += 128) {
    int row = c >> 2;
    int seg = c & 3;
    const bf16_t* gp = W + (size_t)row * H_ + kbase + seg * 8;
    bf16_t* lp = tile + row * 32 + seg * 8;
#if USE_ASYNC_LDS
    __builtin_amdgcn_global_load_async_to_lds_b128((AS1 vi4*)gp, (AS3 vi4*)lp,
                                                   0, 0);
#else
    *(u32x4*)lp = *(const u32x4*)gp;
#endif
  }
}

__device__ __forceinline__ void async_commit_wait() {
#if USE_ASYNC_LDS
  __builtin_amdgcn_s_wait_asynccnt(0);
#endif
  __syncthreads();
}

// ---- weight fp32 -> bf16 -------------------------------------------------
__global__ __launch_bounds__(256)
void cvt_f32_bf16_kernel(const float* __restrict__ src, bf16_t* __restrict__ dst,
                         int n) {
  int i = (blockIdx.x * 256 + threadIdx.x) * 4;
  if (i + 3 < n) {
    f32x4 x = *(const f32x4*)(src + i);
    dst[i + 0] = (bf16_t)x.x;
    dst[i + 1] = (bf16_t)x.y;
    dst[i + 2] = (bf16_t)x.z;
    dst[i + 3] = (bf16_t)x.w;
  }
}

// ---- QKV projection GEMM: out = X @ W^T + bias ---------------------------
// X: [B*S, H] f32 (converted on the fly). W: [H, H] bf16 (row n = out col n).
// mode 0: store [B,NH,S,DK] bf16 (Q,K). mode 1: store [B,NH,DK,S] bf16 (V^T).
// Wave computes 32(M) x 64(N): 8 WMMAs per K-step, B frags from a double-
// buffered LDS tile shared by the whole workgroup (4 waves = 128 M rows).
__global__ __launch_bounds__(128)
void qkv_gemm_kernel(const float* __restrict__ X, const bf16_t* __restrict__ W,
                     const float* __restrict__ bias, bf16_t* __restrict__ out,
                     int mode) {
  __shared__ bf16_t tileB[2][64 * 32];

  const int tid   = threadIdx.x;
  const int lane  = tid & 31;
  const int wave  = tid >> 5;
  const int nbase = blockIdx.x * 64;
  const int mbase = (blockIdx.y * 4 + wave) * 32;

  const bf16_t* Wn = W + (size_t)nbase * H_;

  const v8f z = {0.f,0.f,0.f,0.f,0.f,0.f,0.f,0.f};
  v8f acc0[4], acc1[4];
#pragma unroll
  for (int t = 0; t < 4; ++t) { acc0[t] = z; acc1[t] = z; }

  const float* Xrow0 = X + (size_t)mbase * H_;
  const float* Xrow1 = X + (size_t)(mbase + 16) * H_;

  stage_w_tile(Wn, 0, tileB[0], tid);
  async_commit_wait();

  int buf = 0;
  for (int k = 0; k < H_; k += 32) {
    if (k + 32 < H_) stage_w_tile(Wn, k + 32, tileB[buf ^ 1], tid);
    v16bf a0 = load_matA_f32(Xrow0, H_, k, lane);
    v16bf a1 = load_matA_f32(Xrow1, H_, k, lane);
#pragma unroll
    for (int t = 0; t < 4; ++t) {
      v16bf b = load_matB_lds(&tileB[buf][t * 16 * 32], lane);
      acc0[t] = wmma_bf16(a0, b, acc0[t]);
      acc1[t] = wmma_bf16(a1, b, acc1[t]);
    }
    async_commit_wait();
    buf ^= 1;
  }

  const int hi  = lane >> 4;
  const int col = lane & 15;
#pragma unroll
  for (int t = 0; t < 4; ++t) {
    int n = nbase + t * 16 + col;
    float bv = bias[n];
    int h = n >> 6;          // n / DK
    int d = n & 63;          // n % DK
#pragma unroll
    for (int g = 0; g < 2; ++g) {
      const v8f& a = g ? acc1[t] : acc0[t];
#pragma unroll
      for (int i = 0; i < 8; ++i) {
        int m = mbase + g * 16 + i + hi * 8;
        int b = m >> 11;     // m / S
        int s = m & 2047;    // m % S
        float v = a[i] + bv;
        size_t addr;
        if (mode == 0) addr = ((size_t)(b * NH_ + h) * S_ + s) * DK_ + d;
        else           addr = ((size_t)(b * NH_ + h) * DK_ + d) * S_ + s;
        out[addr] = (bf16_t)v;
      }
    }
  }
}

// ---- flash attention -----------------------------------------------------
// grid: (S/64, B*NH). One wave owns a 16-row query block; loops over keys in
// blocks of 32: 4 WMMAs for Q·K^T, online softmax (16-lane shuffle
// reductions), P through LDS (C layout -> A layout), 4 WMMAs for P·V.
__global__ __launch_bounds__(128)
void flash_attn_kernel(const bf16_t* __restrict__ QH, const bf16_t* __restrict__ KH,
                       const bf16_t* __restrict__ VT, const unsigned char* __restrict__ mask,
                       bf16_t* __restrict__ OB) {
  __shared__ bf16_t Pt[4][16 * 32];

  const int lane  = threadIdx.x & 31;
  const int wave  = threadIdx.x >> 5;
  const int bh    = blockIdx.y;       // b*NH + h
  const int b     = bh >> 4;
  const int h     = bh & 15;
  const int qbase = (blockIdx.x * 4 + wave) * 16;

  const bf16_t* Qb = QH + (size_t)bh * S_ * DK_;
  const bf16_t* Kb = KH + (size_t)bh * S_ * DK_;
  const bf16_t* Vb = VT + (size_t)bh * DK_ * S_;
  const unsigned char* Mb = mask + (size_t)b * S_ * S_;

  // Q fragments (persistent): 16 rows x DK=64 -> two 16x32 A-frags.
  v16bf aq0 = load_matA_bf16(Qb + (size_t)qbase * DK_, DK_, 0,  lane);
  v16bf aq1 = load_matA_bf16(Qb + (size_t)qbase * DK_, DK_, 32, lane);

  const v8f z = {0.f,0.f,0.f,0.f,0.f,0.f,0.f,0.f};
  v8f acc[4];
#pragma unroll
  for (int t = 0; t < 4; ++t) acc[t] = z;

  float row_max[8], row_sum[8];
#pragma unroll
  for (int i = 0; i < 8; ++i) { row_max[i] = NEG_BIG; row_sum[i] = 0.f; }

  const int hi  = lane >> 4;
  const int col = lane & 15;

  for (int j = 0; j < S_; j += 32) {
    // ---- scores: c0 = cols j..j+15, c1 = cols j+16..j+31 ----
    v8f c0 = z, c1 = z;
    c0 = wmma_bf16(aq0, load_matB_bf16(Kb + (size_t)j * DK_, DK_, 0,  lane), c0);
    c0 = wmma_bf16(aq1, load_matB_bf16(Kb + (size_t)j * DK_, DK_, 32, lane), c0);
    c1 = wmma_bf16(aq0, load_matB_bf16(Kb + (size_t)(j + 16) * DK_, DK_, 0,  lane), c1);
    c1 = wmma_bf16(aq1, load_matB_bf16(Kb + (size_t)(j + 16) * DK_, DK_, 32, lane), c1);

    // ---- mask + online softmax ----
    float s0[8], s1[8], mnew[8];
    bool  k0[8], k1[8];
#pragma unroll
    for (int i = 0; i < 8; ++i) {
      int sq = qbase + i + hi * 8;
      const unsigned char* mrow = Mb + (size_t)sq * S_ + j;
      k0[i] = mrow[col] != 0;
      k1[i] = mrow[16 + col] != 0;
      float v0 = k0[i] ? NEG_BIG : c0[i] * 0.125f;   // 1/sqrt(64)
      float v1 = k1[i] ? NEG_BIG : c1[i] * 0.125f;
      s0[i] = v0; s1[i] = v1;
      float mx = redmax16(fmaxf(v0, v1));
      mnew[i] = fmaxf(row_max[i], mx);
    }
#pragma unroll
    for (int i = 0; i < 8; ++i) {
      float corr = __expf(row_max[i] - mnew[i]);     // finite: NEG_BIG sentinel
      row_sum[i] *= corr;
      acc[0][i] *= corr; acc[1][i] *= corr;
      acc[2][i] *= corr; acc[3][i] *= corr;
      float e0 = k0[i] ? 0.f : __expf(s0[i] - mnew[i]);
      float e1 = k1[i] ? 0.f : __expf(s1[i] - mnew[i]);
      row_sum[i] += redsum16(e0 + e1);
      row_max[i] = mnew[i];
      int r = i + hi * 8;
      Pt[wave][r * 32 + col]      = (bf16_t)e0;
      Pt[wave][r * 32 + 16 + col] = (bf16_t)e1;
    }

    // ---- P (A-frag from LDS) x V^T ----
    {
      int r   = lane & 15;
      int klo = hi << 3;
      FragBF fa;
      fa.u[0] = *(const u32x4*)&Pt[wave][r * 32 + klo];
      fa.u[1] = *(const u32x4*)&Pt[wave][r * 32 + klo + 16];
#pragma unroll
      for (int t = 0; t < 4; ++t) {
        v16bf bv = load_matB_bf16(Vb + (size_t)(t * 16) * S_, S_, j, lane);
        acc[t] = wmma_bf16(fa.v, bv, acc[t]);
      }
    }
  }

  // ---- normalize + store attention output [B,S,H] bf16 ----
#pragma unroll
  for (int i = 0; i < 8; ++i) {
    int sq = qbase + i + hi * 8;
    float inv = row_sum[i] > 0.f ? 1.f / row_sum[i] : 0.f;  // all-masked row -> 0
    size_t rowa = ((size_t)(b * S_ + sq)) * H_ + h * DK_;
#pragma unroll
    for (int t = 0; t < 4; ++t)
      OB[rowa + t * 16 + col] = (bf16_t)(acc[t][i] * inv);
  }
}

// ---- output projection: out = OB @ Wo^T + bo (f32 out) -------------------
__global__ __launch_bounds__(128)
void out_proj_kernel(const bf16_t* __restrict__ Xb, const bf16_t* __restrict__ W,
                     const float* __restrict__ bias, float* __restrict__ out) {
  __shared__ bf16_t tileB[2][64 * 32];

  const int tid   = threadIdx.x;
  const int lane  = tid & 31;
  const int wave  = tid >> 5;
  const int nbase = blockIdx.x * 64;
  const int mbase = (blockIdx.y * 4 + wave) * 32;

  const bf16_t* Wn = W + (size_t)nbase * H_;

  const v8f z = {0.f,0.f,0.f,0.f,0.f,0.f,0.f,0.f};
  v8f acc0[4], acc1[4];
#pragma unroll
  for (int t = 0; t < 4; ++t) { acc0[t] = z; acc1[t] = z; }

  const bf16_t* Xrow0 = Xb + (size_t)mbase * H_;
  const bf16_t* Xrow1 = Xb + (size_t)(mbase + 16) * H_;

  stage_w_tile(Wn, 0, tileB[0], tid);
  async_commit_wait();

  int buf = 0;
  for (int k = 0; k < H_; k += 32) {
    if (k + 32 < H_) stage_w_tile(Wn, k + 32, tileB[buf ^ 1], tid);
    v16bf a0 = load_matA_bf16(Xrow0, H_, k, lane);
    v16bf a1 = load_matA_bf16(Xrow1, H_, k, lane);
#pragma unroll
    for (int t = 0; t < 4; ++t) {
      v16bf b = load_matB_lds(&tileB[buf][t * 16 * 32], lane);
      acc0[t] = wmma_bf16(a0, b, acc0[t]);
      acc1[t] = wmma_bf16(a1, b, acc1[t]);
    }
    async_commit_wait();
    buf ^= 1;
  }

  const int hi  = lane >> 4;
  const int col = lane & 15;
#pragma unroll
  for (int t = 0; t < 4; ++t) {
    int n = nbase + t * 16 + col;
    float bv = bias[n];
#pragma unroll
    for (int g = 0; g < 2; ++g) {
      const v8f& a = g ? acc1[t] : acc0[t];
#pragma unroll
      for (int i = 0; i < 8; ++i) {
        int m = mbase + g * 16 + i + hi * 8;
        out[(size_t)m * H_ + n] = a[i] + bv;
      }
    }
  }
}

// ---- host launch ---------------------------------------------------------
extern "C" void kernel_launch(void* const* d_in, const int* in_sizes, int n_in,
                              void* d_out, int out_size, void* d_ws, size_t ws_size,
                              hipStream_t stream) {
  const float* query = (const float*)d_in[0];
  const float* key_  = (const float*)d_in[1];
  const float* value = (const float*)d_in[2];
  const unsigned char* mask = (const unsigned char*)d_in[3]; // bool, 1 byte/elem
  const float* Wq = (const float*)d_in[4];
  const float* bq = (const float*)d_in[5];
  const float* Wk = (const float*)d_in[6];
  const float* bk = (const float*)d_in[7];
  const float* Wv = (const float*)d_in[8];
  const float* bv = (const float*)d_in[9];
  const float* Wo = (const float*)d_in[10];
  const float* bo = (const float*)d_in[11];

  // workspace carve (75.5 MB total), 256B aligned
  char* ws = (char*)d_ws;
  size_t off = 0;
  auto carve = [&](size_t bytes) {
    void* p = ws + off;
    off += (bytes + 255) & ~(size_t)255;
    return p;
  };
  const size_t HH  = (size_t)H_ * H_;
  const size_t BSH = (size_t)B_ * S_ * H_;
  bf16_t* Wqb = (bf16_t*)carve(HH * 2);
  bf16_t* Wkb = (bf16_t*)carve(HH * 2);
  bf16_t* Wvb = (bf16_t*)carve(HH * 2);
  bf16_t* Wob = (bf16_t*)carve(HH * 2);
  bf16_t* QH  = (bf16_t*)carve(BSH * 2);   // [B,NH,S,DK]
  bf16_t* KH  = (bf16_t*)carve(BSH * 2);   // [B,NH,S,DK]
  bf16_t* VT  = (bf16_t*)carve(BSH * 2);   // [B,NH,DK,S]
  bf16_t* OB  = (bf16_t*)carve(BSH * 2);   // [B,S,H]
  (void)ws_size; (void)in_sizes; (void)n_in; (void)out_size;

  // 1) weights -> bf16
  {
    dim3 g((unsigned)(HH / (256 * 4))), blk(256);
    cvt_f32_bf16_kernel<<<g, blk, 0, stream>>>(Wq, Wqb, (int)HH);
    cvt_f32_bf16_kernel<<<g, blk, 0, stream>>>(Wk, Wkb, (int)HH);
    cvt_f32_bf16_kernel<<<g, blk, 0, stream>>>(Wv, Wvb, (int)HH);
    cvt_f32_bf16_kernel<<<g, blk, 0, stream>>>(Wo, Wob, (int)HH);
  }

  // 2) Q, K, V projections (WMMA, LDS-staged weights)
  {
    dim3 g(H_ / 64, (B_ * S_) / 128), blk(128);
    qkv_gemm_kernel<<<g, blk, 0, stream>>>(query, Wqb, bq, QH, 0);
    qkv_gemm_kernel<<<g, blk, 0, stream>>>(key_,  Wkb, bk, KH, 0);
    qkv_gemm_kernel<<<g, blk, 0, stream>>>(value, Wvb, bv, VT, 1);
  }

  // 3) flash attention (WMMA)
  {
    dim3 g(S_ / 64, B_ * NH_), blk(128);
    flash_attn_kernel<<<g, blk, 0, stream>>>(QH, KH, VT, mask, OB);
  }

  // 4) output projection (WMMA, f32 out)
  {
    dim3 g(H_ / 64, (B_ * S_) / 128), blk(128);
    out_proj_kernel<<<g, blk, 0, stream>>>(OB, Wob, bo, (float*)d_out);
  }
}
